// AlgebraicBlock_77695958384932
// MI455X (gfx1250) — matrix operand order repaired and checked by reading, not compiled, not run on hardware
//
#include <hip/hip_runtime.h>

// ---------------------------------------------------------------------------
// CDNA5 (gfx1250) transformer block: mr_norm -> QKV -> ALiBi rational-softmax
// attention -> out-proj(+resid) -> mr_norm -> merged FFN -> rational SwiGLU ->
// down-proj(+resid).
// GEMMs: v_wmma_f32_16x16x32_bf16, double-buffered LDS tiles filled with
// global_load_async_to_lds_b128 (ASYNCcnt), fragments via ds_load_b128.
// ---------------------------------------------------------------------------

typedef __attribute__((ext_vector_type(16))) __bf16 v16bf;
typedef __attribute__((ext_vector_type(8)))  __bf16 v8bf;
typedef __attribute__((ext_vector_type(8)))  float  v8f;

#define WMMA_BF16(a, b, c) \
    __builtin_amdgcn_wmma_f32_16x16x32_bf16(false, (a), false, (b), (short)0, (c), false, false)

// A-fragment (16x32 bf16, M x K). p -> &A[(m0 + (lane&15))*lda + k0 + (lane>>4)*8]
// Halves 0..7 = K base..base+7, halves 8..15 = K base+16..base+23 (ISA 7.12.2).
__device__ inline v16bf load_frag_a(const __bf16* p) {
    union { v16bf v; v8bf h[2]; } u;
    u.h[0] = *(const v8bf*)(p);
    u.h[1] = *(const v8bf*)(p + 16);
    return u.v;
}

// B-fragment (32x16 bf16, K x N), K-major source row.
// p -> &W[(n0 + (lane&15))*ldw + k0 + (lane>>4)*8]; halves 0..7 = K base..base+7, 8..15 = K base+16..base+23.
__device__ inline v16bf load_frag_b(const __bf16* p) {
    union { v16bf v; v8bf h[2]; } u;
    u.h[0] = *(const v8bf*)(p);
    u.h[1] = *(const v8bf*)(p + 16);
    return u.v;
}

// ---- CDNA5 async copy: 16B per lane, global -> LDS, tracked by ASYNCcnt ----
__device__ inline void async_copy_b128(unsigned ldsByteOff, const void* gaddr) {
    asm volatile("global_load_async_to_lds_b128 %0, %1, off"
                 :: "v"(ldsByteOff), "v"((unsigned long long)gaddr)
                 : "memory");
}
#define WAIT_ASYNC_LE6() asm volatile("s_wait_asynccnt 0x6" ::: "memory")
#define WAIT_ASYNC_0()   asm volatile("s_wait_asynccnt 0x0" ::: "memory")

// ---------------------------------------------------------------------------
// Generic fp32 -> bf16 conversion (weights)
// ---------------------------------------------------------------------------
__global__ void cvt_bf16_kernel(const float* __restrict__ in,
                                __bf16* __restrict__ out, long n) {
    long i = (long)blockIdx.x * blockDim.x + threadIdx.x;
    if (i < n) out[i] = (__bf16)in[i];
}

// ---------------------------------------------------------------------------
// RationalMR norm: out = x / (mean|x| + eps) * w, cast to bf16. One row/block.
// ---------------------------------------------------------------------------
__global__ __launch_bounds__(256) void mrnorm_kernel(const float* __restrict__ x,
                                                     const float* __restrict__ w,
                                                     __bf16* __restrict__ out, int C) {
    int row = blockIdx.x;
    const float* xr = x + (size_t)row * C;
    __shared__ float red[256];
    float s = 0.f;
    for (int c = threadIdx.x; c < C; c += 256) s += fabsf(xr[c]);
    red[threadIdx.x] = s;
    __syncthreads();
    for (int off = 128; off > 0; off >>= 1) {
        if ((int)threadIdx.x < off) red[threadIdx.x] += red[threadIdx.x + off];
        __syncthreads();
    }
    float inv = 1.0f / (red[0] / (float)C + 1e-6f);
    for (int c = threadIdx.x; c < C; c += 256)
        out[(size_t)row * C + c] = (__bf16)(xr[c] * inv * w[c]);
}

// ---------------------------------------------------------------------------
// bf16 WMMA GEMM with async-LDS double buffering (K-step 64).
//   C[m][n] = (resid ? resid[m][n] : 0) + sum_k A[m][k]*W[n][k]
// A: MxK bf16 row-major, W: NxK bf16 row-major, C fp32.
// Block = 8 waves; block tile 64x128; wave tile 16x64. M%64==0, N%128==0, K%64==0.
// Dynamic LDS: 2 buffers x (A 64x72 + B 128x72) halves = 55296 bytes.
// ---------------------------------------------------------------------------
#define GP 72               // LDS row pitch in halves (64 data + 8 pad)
#define BUF_HALVES 13824    // (64 + 128) * GP
__global__ __launch_bounds__(256) void gemm_bf16_kernel(const __bf16* __restrict__ A,
                                                        const __bf16* __restrict__ W,
                                                        const float* __restrict__ resid,
                                                        float* __restrict__ C,
                                                        int M, int N, int K) {
    extern __shared__ __bf16 smem[];           // dynamic LDS -> offset 0
    const int tid  = threadIdx.x;
    const int lane = tid & 31;
    const int wave = tid >> 5;
    const int half = lane >> 4;
    const int l15  = lane & 15;
    const int mB = blockIdx.y * 64;
    const int nB = blockIdx.x * 128;
    const int mSub = (wave >> 1) * 16;
    const int nSub = (wave & 1) * 64;

    const int nSteps = K >> 6;

    // stage copy: 6 x 16B per thread (A: 512 chunks, B: 1024 chunks; 8 chunks/row)
    auto issue_stage = [&](int step, int buf) {
        const int k0 = step << 6;
        const unsigned baseA = (unsigned)(buf * BUF_HALVES * 2);        // bytes
        const unsigned baseB = baseA + 64u * GP * 2u;
        #pragma unroll
        for (int rr = 0; rr < 2; ++rr) {
            int idx = tid + rr * 256, row = idx >> 3, c = idx & 7;
            async_copy_b128(baseA + (unsigned)((row * GP + c * 8) * 2),
                            A + (size_t)(mB + row) * K + k0 + c * 8);
        }
        #pragma unroll
        for (int rr = 0; rr < 4; ++rr) {
            int idx = tid + rr * 256, row = idx >> 3, c = idx & 7;
            async_copy_b128(baseB + (unsigned)((row * GP + c * 8) * 2),
                            W + (size_t)(nB + row) * K + k0 + c * 8);
        }
    };

    v8f acc[4] = {};
    auto compute = [&](int buf) {
        const __bf16* sA = smem + (size_t)buf * BUF_HALVES;
        const __bf16* sB = sA + 64 * GP;
        #pragma unroll
        for (int kk = 0; kk < 64; kk += 32) {
            // hoist all fragment loads so DS latency overlaps the WMMA chain
            v16bf a  = load_frag_a(sA + (mSub + l15) * GP + kk + half * 8);
            v16bf b0 = load_frag_b(sB + (nSub +  0 + l15) * GP + kk + half * 8);
            v16bf b1 = load_frag_b(sB + (nSub + 16 + l15) * GP + kk + half * 8);
            v16bf b2 = load_frag_b(sB + (nSub + 32 + l15) * GP + kk + half * 8);
            v16bf b3 = load_frag_b(sB + (nSub + 48 + l15) * GP + kk + half * 8);
            acc[0] = WMMA_BF16(a, b0, acc[0]);
            acc[1] = WMMA_BF16(a, b1, acc[1]);
            acc[2] = WMMA_BF16(a, b2, acc[2]);
            acc[3] = WMMA_BF16(a, b3, acc[3]);
        }
    };

    issue_stage(0, 0);
    for (int s = 0; s + 1 < nSteps; ++s) {
        issue_stage(s + 1, (s + 1) & 1);
        WAIT_ASYNC_LE6();                 // stage s resident (next stage in flight)
        __syncthreads();
        compute(s & 1);
        __syncthreads();                  // readers done before buffer reuse
    }
    WAIT_ASYNC_0();
    __syncthreads();
    compute((nSteps - 1) & 1);

    #pragma unroll
    for (int j = 0; j < 4; ++j)
        #pragma unroll
        for (int r = 0; r < 8; ++r) {
            int m = mB + mSub + half * 8 + r;
            int n = nB + nSub + j * 16 + l15;
            float v = acc[j][r];
            if (resid) v += resid[(size_t)m * N + n];
            C[(size_t)m * N + n] = v;
        }
}

// ---------------------------------------------------------------------------
// Split fused QKV (fp32 [T x 3072]) into per-head bf16 buffers:
//   q[h][t][d] (pre-scaled by 1/sqrt(D)), k[h][t][d], vT[h][d][t]
// ---------------------------------------------------------------------------
__global__ void qkv_prep_kernel(const float* __restrict__ qkv,
                                __bf16* __restrict__ q, __bf16* __restrict__ k,
                                __bf16* __restrict__ vT, int T) {
    long idx = (long)blockIdx.x * blockDim.x + threadIdx.x;
    if (idx >= (long)T * 1024) return;
    int t = (int)(idx >> 10);
    int c = (int)(idx & 1023);
    int h = c >> 6, d = c & 63;
    const float* row = qkv + (size_t)t * 3072;
    q [((size_t)h * T + t) * 64 + d] = (__bf16)(row[c] * 0.125f);      // 1/sqrt(64)
    k [((size_t)h * T + t) * 64 + d] = (__bf16)(row[1024 + c]);
    vT[((size_t)h * 64 + d) * T + t] = (__bf16)(row[2048 + c]);
}

// ---------------------------------------------------------------------------
// Attention: one wave per (head, 16-query tile).
//   scores = (q.kT) - slope*|t-s|, causal -> -1e4; p = rsig(s)^4
//   out = (sum_s p * V) / (sum_s p + eps)
// p-tile re-layout C-frag -> A-frag goes through LDS (ds_store_b16/ds_load_b128).
// ---------------------------------------------------------------------------
__global__ __launch_bounds__(32) void attn_kernel(const __bf16* __restrict__ q,
                                                  const __bf16* __restrict__ k,
                                                  const __bf16* __restrict__ vT,
                                                  __bf16* __restrict__ out, int T) {
    const int lane = threadIdx.x & 31;
    const int half = lane >> 4;
    const int l15  = lane & 15;
    const int h  = blockIdx.y;
    const int t0 = blockIdx.x * 16;

    const __bf16* qh = q  + (size_t)h * T * 64;
    const __bf16* kh = k  + (size_t)h * T * 64;
    const __bf16* vh = vT + (size_t)h * 64 * T;
    const float slope = exp2f(-0.5f * (float)(h + 1));   // ALiBi, H=16

    __shared__ __bf16 smemP[16 * 32];

    const __bf16* qRow = qh + (size_t)(t0 + l15) * 64 + half * 8;
    v16bf aq0 = load_frag_a(qRow);
    v16bf aq1 = load_frag_a(qRow + 32);

    v8f acc[4] = {};
    float rsum[8] = {0.f};

    const int nChunk = t0 / 32 + 1;   // covers all unmasked keys
    for (int ch = 0; ch < nChunk; ++ch) {
        const int s0 = ch * 32;
        // ---- two 16x16 score tiles, then rsig^4 + rowsum + stash to LDS ----
        #pragma unroll
        for (int st = 0; st < 2; ++st) {
            const int sc0 = s0 + st * 16;
            const __bf16* kRow = kh + (size_t)(sc0 + l15) * 64 + half * 8;
            v8f s = {};
            s = WMMA_BF16(aq0, load_frag_b(kRow), s);
            s = WMMA_BF16(aq1, load_frag_b(kRow + 32), s);
            const int scol = sc0 + l15;
            #pragma unroll
            for (int r = 0; r < 8; ++r) {
                int trow = t0 + half * 8 + r;
                float val = s[r] - slope * fabsf((float)(trow - scol));
                if (scol > trow) val = -10000.0f;
                float p = 0.5f * (val / (fabsf(val) + 1.0f) + 1.0f);
                p = p * p; p = p * p;                       // rsig^4
                float rs = p;                               // 16-lane row reduce
                rs += __shfl_xor(rs, 1);
                rs += __shfl_xor(rs, 2);
                rs += __shfl_xor(rs, 4);
                rs += __shfl_xor(rs, 8);
                rsum[r] += rs;
                smemP[(half * 8 + r) * 32 + st * 16 + l15] = (__bf16)p;
            }
        }
        __syncthreads();
        // ---- PV: P(16x32) x V(32x64) ----
        v16bf aP = load_frag_a(smemP + l15 * 32 + half * 8);
        #pragma unroll
        for (int j = 0; j < 4; ++j) {
            v16bf bv = load_frag_b(vh + (size_t)(j * 16 + l15) * T + s0 + half * 8);
            acc[j] = WMMA_BF16(aP, bv, acc[j]);
        }
        __syncthreads();
    }

    #pragma unroll
    for (int j = 0; j < 4; ++j)
        #pragma unroll
        for (int r = 0; r < 8; ++r) {
            int trow = t0 + half * 8 + r;
            int col  = h * 64 + j * 16 + l15;
            out[(size_t)trow * 1024 + col] = (__bf16)(acc[j][r] / (rsum[r] + 1e-6f));
        }
}

// ---------------------------------------------------------------------------
// Rational SwiGLU: hidden = gate * rsig(gate) * val  (merged = [gate | val])
// ---------------------------------------------------------------------------
__global__ void swiglu_kernel(const float* __restrict__ merged,
                              __bf16* __restrict__ hidden, int T, int F) {
    long idx = (long)blockIdx.x * blockDim.x + threadIdx.x;
    if (idx >= (long)T * F) return;
    int t = (int)(idx / F);
    int j = (int)(idx % F);
    float g = merged[(size_t)t * 2 * F + j];
    float v = merged[(size_t)t * 2 * F + F + j];
    float sg = 0.5f * (g / (fabsf(g) + 1.0f) + 1.0f);
    hidden[idx] = (__bf16)(g * sg * v);
}

// ---------------------------------------------------------------------------
extern "C" void kernel_launch(void* const* d_in, const int* in_sizes, int n_in,
                              void* d_out, int out_size, void* d_ws, size_t ws_size,
                              hipStream_t stream) {
    const float* x        = (const float*)d_in[0];
    const float* w_qkv    = (const float*)d_in[1];
    const float* w_out    = (const float*)d_in[2];
    const float* w_merged = (const float*)d_in[3];
    const float* w3       = (const float*)d_in[4];
    const float* n1w      = (const float*)d_in[5];
    const float* n2w      = (const float*)d_in[6];
    float* out = (float*)d_out;

    const int T = 2048, Cm = 1024, F = 4096;
    const size_t LDS_BYTES = 2u * BUF_HALVES * 2u;   // 55296

    char* ws = (char*)d_ws;
    size_t off = 0;
    auto carve = [&](size_t bytes) -> void* {
        void* p = ws + off;
        off = (off + bytes + 255) & ~(size_t)255;
        return p;
    };
    __bf16* wqkv_bf = (__bf16*)carve((size_t)3 * Cm * Cm * 2);
    __bf16* wout_bf = (__bf16*)carve((size_t)Cm * Cm * 2);
    __bf16* wm_bf   = (__bf16*)carve((size_t)2 * F * Cm * 2);
    __bf16* w3_bf   = (__bf16*)carve((size_t)Cm * F * 2);
    __bf16* h1_bf   = (__bf16*)carve((size_t)T * Cm * 2);
    float*  qkv_f   = (float*) carve((size_t)T * 3 * Cm * 4);
    __bf16* q_bf    = (__bf16*)carve((size_t)T * Cm * 2);
    __bf16* k_bf    = (__bf16*)carve((size_t)T * Cm * 2);
    __bf16* vT_bf   = (__bf16*)carve((size_t)T * Cm * 2);
    __bf16* attn_bf = (__bf16*)carve((size_t)T * Cm * 2);
    float*  x2_f    = (float*) carve((size_t)T * Cm * 4);
    __bf16* h2_bf   = (__bf16*)carve((size_t)T * Cm * 2);
    float*  merged_f= (float*) carve((size_t)T * 2 * F * 4);
    __bf16* hid_bf  = (__bf16*)carve((size_t)T * F * 2);

    auto cvt = [&](const float* src, __bf16* dst, long n) {
        cvt_bf16_kernel<<<dim3((unsigned)((n + 255) / 256)), dim3(256), 0, stream>>>(src, dst, n);
    };
    // weights -> bf16
    cvt(w_qkv,    wqkv_bf, (long)3 * Cm * Cm);
    cvt(w_out,    wout_bf, (long)Cm * Cm);
    cvt(w_merged, wm_bf,   (long)2 * F * Cm);
    cvt(w3,       w3_bf,   (long)Cm * F);

    // h1 = mr_norm(x, norm1_w)
    mrnorm_kernel<<<dim3(T), dim3(256), 0, stream>>>(x, n1w, h1_bf, Cm);

    // qkv = h1 @ w_qkv^T   [2048 x 3072]
    gemm_bf16_kernel<<<dim3((3 * Cm) / 128, T / 64), dim3(256), LDS_BYTES, stream>>>(
        h1_bf, wqkv_bf, nullptr, qkv_f, T, 3 * Cm, Cm);

    // split q/k/v per head (q pre-scaled, v transposed)
    qkv_prep_kernel<<<dim3((unsigned)(((long)T * Cm + 255) / 256)), dim3(256), 0, stream>>>(
        qkv_f, q_bf, k_bf, vT_bf, T);

    // attention -> attn_bf [2048 x 1024]
    attn_kernel<<<dim3(T / 16, 16), dim3(32), 0, stream>>>(q_bf, k_bf, vT_bf, attn_bf, T);

    // x2 = x + attn @ w_out^T
    gemm_bf16_kernel<<<dim3(Cm / 128, T / 64), dim3(256), LDS_BYTES, stream>>>(
        attn_bf, wout_bf, x, x2_f, T, Cm, Cm);

    // h2 = mr_norm(x2, norm2_w)
    mrnorm_kernel<<<dim3(T), dim3(256), 0, stream>>>(x2_f, n2w, h2_bf, Cm);

    // merged = h2 @ w_merged^T   [2048 x 8192]
    gemm_bf16_kernel<<<dim3((2 * F) / 128, T / 64), dim3(256), LDS_BYTES, stream>>>(
        h2_bf, wm_bf, nullptr, merged_f, T, 2 * F, Cm);

    // hidden = gate * rsig(gate) * val
    swiglu_kernel<<<dim3((unsigned)(((long)T * F + 255) / 256)), dim3(256), 0, stream>>>(
        merged_f, hid_bf, T, F);

    // out = x2 + hidden @ w3^T   [2048 x 1024]
    gemm_bf16_kernel<<<dim3(Cm / 128, T / 64), dim3(256), LDS_BYTES, stream>>>(
        hid_bf, w3_bf, x2_f, out, T, Cm, F);

    (void)in_sizes; (void)n_in; (void)out_size; (void)ws_size;
}
